// MultiHeadAttention_13950053777990
// MI455X (gfx1250) — matrix operand order repaired and checked by reading, not compiled, not run on hardware
//
#include <hip/hip_runtime.h>
#include <hip/hip_bf16.h>

// MI455X / gfx1250 fused multi-head attention block:
//   qh/kh/vh = X @ W^T + b   (WMMA f16 -> f32 acc, stored f16)
//   S = qh kh^T / 8, mask, softmax  -> attns output (f32, 537MB stream)
//   ctx = P @ vh             (WMMA, P requantized f16 from LDS)
//   y = LN(ctx @ Wfc^T + bfc + q)
//
// Bandwidth-bound on the attns write; all GEMMs on v_wmma_f32_16x16x32_f16.
// LDS tiles are stored so every WMMA fragment is assembled from contiguous
// 16B chunks (ds_load_b128 / global_load_b128), not scalar u16 loads.

typedef __attribute__((ext_vector_type(16))) _Float16 v16h;
typedef __attribute__((ext_vector_type(8)))  _Float16 v8h;
typedef __attribute__((ext_vector_type(8)))  float    v8f;

#define B_   4
#define L_   2048
#define E_   512
#define H_   8
#define DK_  64
#define HD_  512          // H*DK
#define M_   (B_ * L_)    // 8192 rows

__device__ __forceinline__ v8f wmma16(v16h a, v16h b, v8f c) {
  // (neg_a, A, neg_b, B, c_mod, C, reuse_a, reuse_b)
  return __builtin_amdgcn_wmma_f32_16x16x32_f16(false, a, false, b, (short)0, c,
                                                false, false);
}

// A-matrix fragment (16x32 f16, ISA 7.12.2): lane half lh, row base points at
// the lane's 32-halfword row slice.  e 0..15 -> K = lh*16+e (contiguous,
// matching the B fragment) => two 16B vector loads.
__device__ __forceinline__ v16h load_a_frag(const _Float16* rowbase, int lh) {
  v8h lo = *(const v8h*)(rowbase + lh * 16);
  v8h hi = *(const v8h*)(rowbase + lh * 16 + 8);
  return __builtin_shufflevector(lo, hi, 0, 1, 2, 3, 4, 5, 6, 7,
                                         8, 9, 10, 11, 12, 13, 14, 15);
}

// B-matrix fragment (32x16 f16): lane's column fixed, K = lh*16 + e,
// 16 contiguous halfwords = one 32B vector load.
__device__ __forceinline__ v16h load_b_frag(const _Float16* p) {
  return *(const v16h*)p;
}

// CDNA5 async global->LDS copy, 16 bytes/lane (ASYNCcnt tracked).
__device__ __forceinline__ void async_g2l_b128(unsigned lds_addr, unsigned goff,
                                               const void* gbase) {
  asm volatile("global_load_async_to_lds_b128 %0, %1, %2"
               :: "v"(lds_addr), "v"(goff), "s"(gbase) : "memory");
}
__device__ __forceinline__ void wait_async0() {
  asm volatile("s_wait_asynccnt 0" ::: "memory");
}
__device__ __forceinline__ unsigned lds_off(const void* p) {
  return (unsigned)(unsigned long long)p;   // low 32 bits = LDS byte address
}

// ---------------------------------------------------------------------------
// Kernel 1: Y[M,512] = (f16) (X[M,512] @ W[512,512]^T + bias)
// block tile 128x64, 8 waves as 4(M) x 2(N), each wave: 2x2 of 16x16 WMMA.
// A tile [m][k], B tile k-major per column: sBt[n][k].
// ---------------------------------------------------------------------------
__global__ __launch_bounds__(256) void proj_gemm(
    const float* __restrict__ X, const float* __restrict__ W,
    const float* __restrict__ bias, _Float16* __restrict__ Y) {
  __shared__ _Float16 sA[128 * 32];
  __shared__ _Float16 sBt[64 * 32];

  const int K = E_, N = HD_;
  const int tid  = threadIdx.x;
  const int lane = tid & 31, wave = tid >> 5;
  const int lh = lane >> 4, lr = lane & 15;
  const int waveM = wave & 3, waveN = wave >> 2;
  const int m0 = blockIdx.y * 128, n0 = blockIdx.x * 64;

  v8f acc[2][2] = {};

  for (int k0 = 0; k0 < K; k0 += 32) {
    if (k0 + 32 < K) {  // global_prefetch_b8 for the next K tile
      __builtin_prefetch(&X[(size_t)(m0 + (tid >> 1)) * K + k0 + 32], 0, 0);
      __builtin_prefetch(&W[(size_t)(n0 + (tid & 63)) * K + k0 + 32], 0, 0);
    }
    for (int i = tid; i < 128 * 32; i += 256) {   // A tile f32->f16
      int mm = i >> 5, kk = i & 31;
      sA[i] = (_Float16)X[(size_t)(m0 + mm) * K + k0 + kk];
    }
    for (int i = tid; i < 64 * 32; i += 256) {    // B tile = W slice, [n][k]
      int nn = i >> 5, kk = i & 31;
      sBt[i] = (_Float16)W[(size_t)(n0 + nn) * K + k0 + kk];
    }
    __syncthreads();

    v16h afr[2];
#pragma unroll
    for (int sm = 0; sm < 2; ++sm)
      afr[sm] = load_a_frag(&sA[(waveM * 32 + sm * 16 + lr) * 32], lh);
#pragma unroll
    for (int sn = 0; sn < 2; ++sn) {
      int nloc = waveN * 32 + sn * 16 + lr;
      v16h bfr = load_b_frag(&sBt[nloc * 32 + lh * 16]);
#pragma unroll
      for (int sm = 0; sm < 2; ++sm)
        acc[sm][sn] = wmma16(afr[sm], bfr, acc[sm][sn]);
    }
    __syncthreads();
  }

#pragma unroll
  for (int sm = 0; sm < 2; ++sm)
#pragma unroll
    for (int sn = 0; sn < 2; ++sn) {
      int mbase = m0 + waveM * 32 + sm * 16 + lh * 8;
      int n = n0 + waveN * 32 + sn * 16 + lr;
      float bn = bias[n];
#pragma unroll
      for (int r = 0; r < 8; ++r)
        Y[(size_t)(mbase + r) * N + n] = (_Float16)(acc[sm][sn][r] + bn);
    }
}

// ---------------------------------------------------------------------------
// Kernel 2: attention. One block = (batch, head, 16-query tile).
// Dynamic LDS: 16x2048 f32 score/prob buffer (128KB) + 16x64 f16 Q tile.
// ---------------------------------------------------------------------------
__global__ __launch_bounds__(256) void attn_kernel(
    const _Float16* __restrict__ qh, const _Float16* __restrict__ kh,
    const _Float16* __restrict__ vh, const unsigned char* __restrict__ mask,
    float* __restrict__ attn_out, _Float16* __restrict__ ctx) {
  extern __shared__ char smem[];
  float*    sS = (float*)smem;                          // 16 x 2048 f32
  _Float16* sQ = (_Float16*)(smem + 16 * L_ * 4);       // 16 x 64 f16

  const int bh = blockIdx.y;
  const int bb = bh >> 3, h = bh & 7;
  const int q0 = blockIdx.x * 16;
  const int tid  = threadIdx.x;
  const int lane = tid & 31, wave = tid >> 5;
  const int lh = lane >> 4, lr = lane & 15;

  // ---- Q tile: async global->LDS, 2048B = 128 lanes x 16B ----
  if (tid < 128) {
    int mm = tid >> 3, ch = tid & 7;
    unsigned goff = (unsigned)((((bb * L_ + q0 + mm) * HD_) + h * DK_ + ch * 8) * 2);
    async_g2l_b128(lds_off(sQ) + (unsigned)((mm * 64 + ch * 8) * 2), goff, qh);
  }
  wait_async0();
  __syncthreads();

  v16h aq[2];
#pragma unroll
  for (int s = 0; s < 2; ++s)
    aq[s] = load_a_frag(&sQ[lr * 64 + s * 32], lh);

  // ---- scores: S[16, 2048] = Q K^T / sqrt(64), mask -> LDS ----
  for (int j = wave; j < L_ / 16; j += 8) {
    int key0 = j * 16;
    v8f c = {};
#pragma unroll
    for (int s = 0; s < 2; ++s) {
      // B = K^T: lane's key row, K-dim slice s*32 + lh*16 .. +15 contiguous
      v16h bfr = load_b_frag(&kh[(size_t)(bb * L_ + key0 + lr) * HD_ +
                                 h * DK_ + s * 32 + lh * 16]);
      c = wmma16(aq[s], bfr, c);
    }
#pragma unroll
    for (int r = 0; r < 8; ++r) {
      int m = lh * 8 + r, kk = key0 + lr;
      float sv = c[r] * 0.125f;
      if (mask[(size_t)(bb * L_ + q0 + m) * L_ + kk]) sv = -1e9f;
      sS[m * L_ + kk] = sv;
    }
  }
  __syncthreads();

  // ---- softmax over each of the 16 rows (2 rows / wave, wave32 shuffles) ----
#pragma unroll
  for (int rr = 0; rr < 2; ++rr) {
    int m = wave * 2 + rr;
    float mx = -3.4e38f;
    for (int kk = lane; kk < L_; kk += 32) mx = fmaxf(mx, sS[m * L_ + kk]);
#pragma unroll
    for (int off = 16; off > 0; off >>= 1) mx = fmaxf(mx, __shfl_xor(mx, off, 32));
    float sum = 0.f;
    for (int kk = lane; kk < L_; kk += 32) {
      float ev = __expf(sS[m * L_ + kk] - mx);
      sS[m * L_ + kk] = ev;
      sum += ev;
    }
#pragma unroll
    for (int off = 16; off > 0; off >>= 1) sum += __shfl_xor(sum, off, 32);
    float inv = 1.0f / sum;
    // normalize in LDS + stream the f32 probabilities out (the 537MB output)
    for (int kk = lane; kk < L_; kk += 32) {
      float p = sS[m * L_ + kk] * inv;
      sS[m * L_ + kk] = p;
      attn_out[((size_t)(h * B_ + bb) * L_ + (q0 + m)) * L_ + kk] = p;
    }
  }
  __syncthreads();

  // ---- ctx = P[16,2048] @ V[2048,64]; K split over 8 waves (8 chunks each) --
  v8f oc[4] = {};
  for (int kc = wave * 8; kc < wave * 8 + 8; ++kc) {
    // A = P slice, f32 in LDS -> f16 (contiguous 8-float groups vectorize)
    const float* pa = &sS[lr * L_ + kc * 32];
    v16h ap;
#pragma unroll
    for (int e = 0; e < 8; ++e) ap[e] = (_Float16)pa[lh * 16 + e];
#pragma unroll
    for (int e = 0; e < 8; ++e) ap[8 + e] = (_Float16)pa[lh * 16 + 8 + e];
#pragma unroll
    for (int t = 0; t < 4; ++t) {
      v16h bv;
#pragma unroll
      for (int e = 0; e < 16; ++e)
        bv[e] = vh[(size_t)(bb * L_ + kc * 32 + lh * 16 + e) * HD_ +
                   h * DK_ + t * 16 + lr];
      oc[t] = wmma16(ap, bv, oc[t]);
    }
  }
  __syncthreads();   // done reading probabilities; reuse sS for reduction

  float* sR = sS;    // [wave][t][m][n] : 8*4*16*16 f32 = 32KB
#pragma unroll
  for (int t = 0; t < 4; ++t)
#pragma unroll
    for (int r = 0; r < 8; ++r)
      sR[((wave * 4 + t) * 16 + (lh * 8 + r)) * 16 + lr] = oc[t][r];
  __syncthreads();

  for (int i = tid; i < 16 * 64; i += 256) {
    int m = i >> 6, n = i & 63, t = n >> 4, nn = n & 15;
    float s = 0.f;
#pragma unroll
    for (int w = 0; w < 8; ++w) s += sR[((w * 4 + t) * 16 + m) * 16 + nn];
    ctx[(size_t)(bb * L_ + q0 + m) * HD_ + h * DK_ + n] = (_Float16)s;
  }
}

// ---------------------------------------------------------------------------
// Kernel 3: X[M,512] = ctx @ Wfc^T + bfc + residual   (f32 out)
// A tile is already f16 -> async b128 copy into LDS.
// ---------------------------------------------------------------------------
__global__ __launch_bounds__(256) void fc_gemm(
    const _Float16* __restrict__ Xh, const float* __restrict__ W,
    const float* __restrict__ bias, const float* __restrict__ resid,
    float* __restrict__ Y) {
  __shared__ _Float16 sA[128 * 32];
  __shared__ _Float16 sBt[64 * 32];

  const int K = HD_, N = E_;
  const int tid  = threadIdx.x;
  const int lane = tid & 31, wave = tid >> 5;
  const int lh = lane >> 4, lr = lane & 15;
  const int waveM = wave & 3, waveN = wave >> 2;
  const int m0 = blockIdx.y * 128, n0 = blockIdx.x * 64;

  v8f acc[2][2] = {};

  for (int k0 = 0; k0 < K; k0 += 32) {
    // A tile: 128x32 f16 = 8192B as 512 x 16B async chunks (2 per thread)
    for (int c0 = tid; c0 < 512; c0 += 256) {
      int mm = c0 >> 2, ch = c0 & 3;
      unsigned goff = (unsigned)((((m0 + mm) * K) + k0 + ch * 8) * 2);
      async_g2l_b128(lds_off(sA) + (unsigned)((mm * 32 + ch * 8) * 2), goff, Xh);
    }
    for (int i = tid; i < 64 * 32; i += 256) {    // B tile f32->f16, [n][k]
      int nn = i >> 5, kk = i & 31;
      sBt[i] = (_Float16)W[(size_t)(n0 + nn) * K + k0 + kk];
    }
    wait_async0();
    __syncthreads();

    v16h afr[2];
#pragma unroll
    for (int sm = 0; sm < 2; ++sm)
      afr[sm] = load_a_frag(&sA[(waveM * 32 + sm * 16 + lr) * 32], lh);
#pragma unroll
    for (int sn = 0; sn < 2; ++sn) {
      int nloc = waveN * 32 + sn * 16 + lr;
      v16h bfr = load_b_frag(&sBt[nloc * 32 + lh * 16]);
#pragma unroll
      for (int sm = 0; sm < 2; ++sm)
        acc[sm][sn] = wmma16(afr[sm], bfr, acc[sm][sn]);
    }
    __syncthreads();
  }

#pragma unroll
  for (int sm = 0; sm < 2; ++sm)
#pragma unroll
    for (int sn = 0; sn < 2; ++sn) {
      int mbase = m0 + waveM * 32 + sm * 16 + lh * 8;
      int n = n0 + waveN * 32 + sn * 16 + lr;
      float bn = bias[n];
#pragma unroll
      for (int r = 0; r < 8; ++r) {
        size_t idx = (size_t)(mbase + r) * N + n;
        Y[idx] = acc[sm][sn][r] + bn + resid[idx];
      }
    }
}

// ---------------------------------------------------------------------------
// Kernel 4: row-wise LayerNorm over 512 columns; one block per row.
// ---------------------------------------------------------------------------
__global__ __launch_bounds__(256) void ln_kernel(
    const float* __restrict__ x, const float* __restrict__ gamma,
    const float* __restrict__ beta, float* __restrict__ out) {
  __shared__ float red[8], red2[8];
  const int row = blockIdx.x;
  const int tid = threadIdx.x, lane = tid & 31, wave = tid >> 5;
  const float* xr = x + (size_t)row * E_;

  float v0 = xr[tid], v1 = xr[tid + 256];
  float s = v0 + v1;
#pragma unroll
  for (int off = 16; off > 0; off >>= 1) s += __shfl_xor(s, off, 32);
  if (lane == 0) red[wave] = s;
  __syncthreads();
  float tot = 0.f;
#pragma unroll
  for (int w = 0; w < 8; ++w) tot += red[w];
  float mean = tot * (1.0f / E_);

  float d0 = v0 - mean, d1 = v1 - mean;
  float ss = d0 * d0 + d1 * d1;
#pragma unroll
  for (int off = 16; off > 0; off >>= 1) ss += __shfl_xor(ss, off, 32);
  if (lane == 0) red2[wave] = ss;
  __syncthreads();
  float tot2 = 0.f;
#pragma unroll
  for (int w = 0; w < 8; ++w) tot2 += red2[w];
  float inv = rsqrtf(tot2 * (1.0f / E_) + 1e-5f);

  out[(size_t)row * E_ + tid]       = d0 * inv * gamma[tid] + beta[tid];
  out[(size_t)row * E_ + tid + 256] = d1 * inv * gamma[tid + 256] + beta[tid + 256];
}

// ---------------------------------------------------------------------------
extern "C" void kernel_launch(void* const* d_in, const int* in_sizes, int n_in,
                              void* d_out, int out_size, void* d_ws, size_t ws_size,
                              hipStream_t stream) {
  (void)in_sizes; (void)n_in; (void)out_size; (void)ws_size;
  const float* q    = (const float*)d_in[0];
  const float* k    = (const float*)d_in[1];
  const float* v    = (const float*)d_in[2];
  const unsigned char* mask = (const unsigned char*)d_in[3];
  const float* Wq   = (const float*)d_in[4];
  const float* bq   = (const float*)d_in[5];
  const float* Wk   = (const float*)d_in[6];
  const float* bk   = (const float*)d_in[7];
  const float* Wv   = (const float*)d_in[8];
  const float* bv   = (const float*)d_in[9];
  const float* Wfc  = (const float*)d_in[10];
  const float* bfc  = (const float*)d_in[11];
  const float* gmm  = (const float*)d_in[12];
  const float* bta  = (const float*)d_in[13];

  float* outputs = (float*)d_out;                           // [B,L,E] f32
  float* attns   = outputs + (size_t)B_ * L_ * E_;          // [H*B,L,L] f32

  // workspace: qh | kh | vh | ctx (f16, 8MB each) | xbuf (f32, 16MB)
  _Float16* qh  = (_Float16*)d_ws;
  _Float16* kh  = qh + (size_t)M_ * HD_;
  _Float16* vh  = kh + (size_t)M_ * HD_;
  _Float16* ctx = vh + (size_t)M_ * HD_;
  float*    xbuf = (float*)(ctx + (size_t)M_ * HD_);

  dim3 gGemm(HD_ / 64, M_ / 128);                 // (8, 64)
  proj_gemm<<<gGemm, 256, 0, stream>>>(q, Wq, bq, qh);
  proj_gemm<<<gGemm, 256, 0, stream>>>(k, Wk, bk, kh);
  proj_gemm<<<gGemm, 256, 0, stream>>>(v, Wv, bv, vh);

  dim3 gAttn(L_ / 16, B_ * H_);                   // (128, 32)
  size_t smem = (size_t)16 * L_ * 4 + (size_t)16 * 64 * 2;  // 130KB dynamic LDS
  attn_kernel<<<gAttn, 256, smem, stream>>>(qh, kh, vh, mask, attns, ctx);

  fc_gemm<<<gGemm, 256, 0, stream>>>(ctx, Wfc, bfc, q, xbuf);

  ln_kernel<<<dim3(M_), 256, 0, stream>>>(xbuf, gmm, bta, outputs);
}
